// GCNContextPaper_76948634075448
// MI455X (gfx1250) — matrix operand order repaired and checked by reading, not compiled, not run on hardware
//
#include <hip/hip_runtime.h>
#include <math.h>

#define USE_ASYNC_LDS 1   // gfx1250 GLOBAL_LOAD_ASYNC_TO_LDS_B128 staging path

// ---------------------------------------------------------------------------
// Types for WMMA fragments (gfx1250, wave32)
// ---------------------------------------------------------------------------
typedef __attribute__((ext_vector_type(16))) __bf16 v16bf;
typedef __attribute__((ext_vector_type(8)))  float  v8f;

struct alignas(16) U4 { unsigned int x, y, z, w; };

union FragBF {
    v16bf v;
    U4    q[2];
    __device__ FragBF() {}
};

__device__ __forceinline__ unsigned short f2bf(float f) {
    union { __bf16 h; unsigned short u; } cv;
    cv.h = (__bf16)f;            // hardware fptrunc f32 -> bf16
    return cv.u;
}

__device__ __forceinline__ float gelu_exact(float v) {
    return 0.5f * v * (1.0f + erff(v * 0.70710678118654752f));
}

#if USE_ASYNC_LDS
// Per-lane async DMA: LDS[lds_byte_off] <= 16B at gaddr. Tracked by ASYNCcnt.
__device__ __forceinline__ void async_copy_b128(unsigned lds_byte_off, const void* gaddr) {
    asm volatile("global_load_async_to_lds_b128 %0, %1, off"
                 :: "v"(lds_byte_off), "v"((unsigned long long)(uintptr_t)gaddr)
                 : "memory");
}
#endif

// ---------------------------------------------------------------------------
// Tiled GEMM on pre-converted bf16 operands:
//   C[M,N] = A[M,K] @ Bt[N,K]^T   (Bt row-major [n][k] = B transposed)
// BM=128, BN=128, BK=32; 256 threads = 8 waves (4m x 2n); 8 WMMA / K-step /
// wave; double-buffered LDS with async global->LDS staging.
// Epilogue (optional): +bias, exact GELU, fp32 C, bf16 Cb, C2 = v*dis[row]^2.
// Requires N % 128 == 0, K % 32 == 0 (all calls comply); M clamped/guarded.
// ---------------------------------------------------------------------------
#define BM 128
#define BN 128
#define BK 32
// dynamic LDS layout (bytes): A0 | A1 | B0 | B1, 8 KB each
#define LDS_A(buf) ((unsigned)(buf) * 8192u)
#define LDS_B(buf) (16384u + (unsigned)(buf) * 8192u)

__global__ __launch_bounds__(256) void gemm_bf16_wmma(
    const unsigned short* __restrict__ A,  int lda,   // bf16 [M][K]
    const unsigned short* __restrict__ Bt, int ldb,   // bf16 [N][K]
    float*          __restrict__ C,   int ldc,        // nullable fp32 out
    unsigned short* __restrict__ Cb,  int ldcb,       // nullable bf16 out
    float*          __restrict__ C2,  int ldc2,       // nullable fp32 out * dis^2
    const float*    __restrict__ dis,                 // row scale for C2
    const float*    __restrict__ bias,
    int M, int N, int K, int fuse_gelu)
{
    extern __shared__ unsigned char smem[];           // 32 KB, dyn offset 0

    const int tid    = threadIdx.x;
    const int lane   = tid & 31;
    const int wave   = tid >> 5;          // 0..7
    const int wave_m = wave >> 1;         // 0..3 -> 32-row strip
    const int wave_n = wave & 1;          // 0..1 -> 64-col strip
    const int bm     = blockIdx.x * BM;
    const int bn     = blockIdx.y * BN;

    const int m_l = lane & 15;
    const int hi  = lane >> 4;

    v8f acc[2][4] = {};

    // staging decomposition: 512 x 16B chunks per tile, 2 per thread.
    // Rows >= M are clamped: their data only feeds output rows that are
    // never stored (WMMA rows are independent), so no masking is needed.
    const int srow = tid >> 2;            // 0..63
    const int sc8  = (tid & 3) << 3;      // halfword offset 0,8,16,24
    int ar0 = bm + srow;       if (ar0 >= M) ar0 = M - 1;
    int ar1 = bm + srow + 64;  if (ar1 >= M) ar1 = M - 1;
    const unsigned short* Ap0 = A  + (size_t)ar0 * lda + sc8;
    const unsigned short* Ap1 = A  + (size_t)ar1 * lda + sc8;
    const unsigned short* Bp0 = Bt + (size_t)(bn + srow)      * ldb + sc8;
    const unsigned short* Bp1 = Bt + (size_t)(bn + srow + 64) * ldb + sc8;
    const unsigned relA0 = (unsigned)(srow        * BK + sc8) * 2u;   // bytes
    const unsigned relA1 = (unsigned)((srow + 64) * BK + sc8) * 2u;

#if USE_ASYNC_LDS
    #define STAGE_TILE(k0, buf)                                           \
        do {                                                              \
            async_copy_b128(LDS_A(buf) + relA0, Ap0 + (k0));              \
            async_copy_b128(LDS_A(buf) + relA1, Ap1 + (k0));              \
            async_copy_b128(LDS_B(buf) + relA0, Bp0 + (k0));              \
            async_copy_b128(LDS_B(buf) + relA1, Bp1 + (k0));              \
        } while (0)
#else
    #define STAGE_TILE(k0, buf)                                           \
        do {                                                              \
            *(U4*)(smem + LDS_A(buf) + relA0) = *(const U4*)(Ap0 + (k0)); \
            *(U4*)(smem + LDS_A(buf) + relA1) = *(const U4*)(Ap1 + (k0)); \
            *(U4*)(smem + LDS_B(buf) + relA0) = *(const U4*)(Bp0 + (k0)); \
            *(U4*)(smem + LDS_B(buf) + relA1) = *(const U4*)(Bp1 + (k0)); \
        } while (0)
#endif

    const int nk = K / BK;
    STAGE_TILE(0, 0);

    for (int i = 0; i < nk; ++i) {
        const int cur = i & 1;
        if (i + 1 < nk) {
            // overwrite of buf cur^1 is safe: all waves finished reading it
            // at the end-of-compute barrier of iteration i-1.
            STAGE_TILE((i + 1) * BK, cur ^ 1);
#if USE_ASYNC_LDS
            asm volatile("s_wait_asynccnt 0x4" ::: "memory");  // tile i done
#endif
        } else {
#if USE_ASYNC_LDS
            asm volatile("s_wait_asynccnt 0x0" ::: "memory");
#endif
        }
        __syncthreads();   // all waves' staging of buf cur complete

        const unsigned short* ldsA = (const unsigned short*)(smem + LDS_A(cur));
        const unsigned short* ldsB = (const unsigned short*)(smem + LDS_B(cur));

        // A fragments (ISA 7.12.2 16-bit A 16x32 layout):
        // lane = hi*16 + m : halves 0..7 -> K=hi*8..+7, halves 8..15 -> +16
        FragBF a[2];
#pragma unroll
        for (int mi = 0; mi < 2; ++mi) {
            const unsigned short* ap =
                &ldsA[(wave_m * 32 + mi * 16 + m_l) * BK + hi * 8];
            a[mi].q[0] = *(const U4*)(ap);
            a[mi].q[1] = *(const U4*)(ap + 16);
        }
        // 4 B fragments; lane = hi*16 + n : halves 0..7 -> K=hi*8..+7, halves 8..15 -> +16
#pragma unroll
        for (int j = 0; j < 4; ++j) {
            FragBF b;
            const unsigned short* bp =
                &ldsB[(wave_n * 64 + j * 16 + m_l) * BK + hi * 8];
            b.q[0] = *(const U4*)(bp);
            b.q[1] = *(const U4*)(bp + 16);
            acc[0][j] = __builtin_amdgcn_wmma_f32_16x16x32_bf16(
                false, a[0].v, false, b.v, (short)0, acc[0][j], false, false);
            acc[1][j] = __builtin_amdgcn_wmma_f32_16x16x32_bf16(
                false, a[1].v, false, b.v, (short)0, acc[1][j], false, false);
        }
        __syncthreads();   // done reading buf cur
    }

    // epilogue: lane holds col n=m_l, rows hi*8 .. hi*8+7 (C/D layout)
    const int mg = hi * 8;
#pragma unroll
    for (int mi = 0; mi < 2; ++mi) {
#pragma unroll
        for (int j = 0; j < 4; ++j) {
            int col = bn + wave_n * 64 + j * 16 + m_l;
            float bv = bias ? bias[col] : 0.0f;
#pragma unroll
            for (int r = 0; r < 8; ++r) {
                int row = bm + wave_m * 32 + mi * 16 + mg + r;
                if (row < M) {
                    float v = acc[mi][j][r] + bv;
                    if (fuse_gelu) v = gelu_exact(v);
                    if (C)  C[(size_t)row * ldc + col] = v;
                    if (Cb) Cb[(size_t)row * ldcb + col] = f2bf(v);
                    if (C2) {
                        float d = dis[row];
                        C2[(size_t)row * ldc2 + col] = v * d * d;
                    }
                }
            }
        }
    }
}

// ---------------------------------------------------------------------------
// One-time (per launch) conversions
// ---------------------------------------------------------------------------
__global__ __launch_bounds__(256) void cvt_f32_bf16_kernel(
    const float* __restrict__ in, unsigned short* __restrict__ out, size_t n4)
{
    size_t i = (size_t)blockIdx.x * blockDim.x + threadIdx.x;   // 4 elems each
    if (i < n4) {
        const float* p = in + i * 4;
        unsigned short o[4];
        o[0] = f2bf(p[0]); o[1] = f2bf(p[1]); o[2] = f2bf(p[2]); o[3] = f2bf(p[3]);
        *(unsigned long long*)(out + i * 4) = *(unsigned long long*)o;
    }
}

// W is [K][Nn] row-major fp32; Bt is [Nn][K] row-major bf16
__global__ __launch_bounds__(256) void transpose_bf16_kernel(
    const float* __restrict__ W, unsigned short* __restrict__ Bt, int K, int Nn)
{
    size_t i = (size_t)blockIdx.x * blockDim.x + threadIdx.x;
    size_t total = (size_t)K * Nn;
    if (i < total) {
        int k = (int)(i / Nn);
        int n = (int)(i % Nn);
        Bt[(size_t)n * K + k] = f2bf(W[i]);
    }
}

// ---------------------------------------------------------------------------
// Degree / normalization
// ---------------------------------------------------------------------------
__global__ void init_deg_kernel(float* deg, int n) {
    int i = blockIdx.x * blockDim.x + threadIdx.x;
    if (i < n) deg[i] = 1.0f;   // self loop
}

__global__ void count_deg_kernel(const int* __restrict__ dst, float* __restrict__ deg, int e) {
    int i = blockIdx.x * blockDim.x + threadIdx.x;
    if (i < e) atomicAdd(&deg[dst[i]], 1.0f);
}

__global__ void make_dis_kernel(const float* __restrict__ deg, float* __restrict__ dis, int n) {
    int i = blockIdx.x * blockDim.x + threadIdx.x;
    if (i < n) {
        float d = deg[i];
        dis[i] = d > 0.0f ? rsqrtf(d) : 0.0f;
    }
}

// ---------------------------------------------------------------------------
// Edge scatter: out[dst] += hw[src] * dis[src] * dis[dst]  (one block / edge)
// ---------------------------------------------------------------------------
__global__ __launch_bounds__(256) void edge_scatter_kernel(
    const int* __restrict__ src, const int* __restrict__ dst,
    const float* __restrict__ dis, const float* __restrict__ hw,
    float* __restrict__ out, int ldo, int e, int h)
{
    int ei = blockIdx.x;
    if (ei >= e) return;
    int s = src[ei], d = dst[ei];
    float nrm = dis[s] * dis[d];
    const float* xs = hw + (size_t)s * h;
    float* xd = out + (size_t)d * ldo;
    for (int c = threadIdx.x; c < h; c += 256)
        atomicAdd(&xd[c], xs[c] * nrm);
}

// ---------------------------------------------------------------------------
// bias + exact GELU on strided fp32 slice -> bf16 mirror slice
// ---------------------------------------------------------------------------
__global__ __launch_bounds__(256) void gelu_bias_kernel(
    const float* __restrict__ X, int ldx,
    unsigned short* __restrict__ Xb, int ldxb,
    const float* __restrict__ bias, int n, int h)
{
    size_t i = (size_t)blockIdx.x * blockDim.x + threadIdx.x;
    size_t total = (size_t)n * h;
    if (i < total) {
        int row = (int)(i / h);
        int c   = (int)(i % h);
        float v = X[(size_t)row * ldx + c] + bias[c];
        Xb[(size_t)row * ldxb + c] = f2bf(gelu_exact(v));
    }
}

// ---------------------------------------------------------------------------
// Row LayerNorm; fp32 out Y and/or bf16 out Yb (either nullable)
// ---------------------------------------------------------------------------
__global__ __launch_bounds__(256) void layernorm_kernel(
    const float* __restrict__ X, int ldx,
    float* __restrict__ Y, int ldy,
    unsigned short* __restrict__ Yb, int ldyb,
    const float* __restrict__ g, const float* __restrict__ b, int h)
{
    int row = blockIdx.x;
    const float* x = X + (size_t)row * ldx;
    float s = 0.f, ss = 0.f;
    for (int c = threadIdx.x; c < h; c += 256) {
        float v = x[c];
        s += v; ss += v * v;
    }
    __shared__ float r1[256], r2[256];
    r1[threadIdx.x] = s; r2[threadIdx.x] = ss;
    __syncthreads();
    for (int w = 128; w > 0; w >>= 1) {
        if (threadIdx.x < w) {
            r1[threadIdx.x] += r1[threadIdx.x + w];
            r2[threadIdx.x] += r2[threadIdx.x + w];
        }
        __syncthreads();
    }
    float mu  = r1[0] / h;
    float var = r2[0] / h - mu * mu;
    float inv = rsqrtf(var + 1e-5f);
    for (int c = threadIdx.x; c < h; c += 256) {
        float v = (x[c] - mu) * inv * g[c] + b[c];
        if (Y)  Y[(size_t)row * ldy + c] = v;
        if (Yb) Yb[(size_t)row * ldyb + c] = f2bf(v);
    }
}

// ---------------------------------------------------------------------------
// Orchestration
// ---------------------------------------------------------------------------
extern "C" void kernel_launch(void* const* d_in, const int* in_sizes, int n_in,
                              void* d_out, int out_size, void* d_ws, size_t ws_size,
                              hipStream_t stream) {
    const int IN_F = 1536, H = 768, HOPS = 3;
    const int CATW = (HOPS + 1) * H;   // 3072

    const float* x      = (const float*)d_in[0];
    const int*   ei     = (const int*)d_in[1];
    const float* proj_w = (const float*)d_in[2];
    const float* proj_b = (const float*)d_in[3];
    const float* ln1_g  = (const float*)d_in[4];
    const float* ln1_b  = (const float*)d_in[5];
    const float* gcn_w  = (const float*)d_in[6];
    const float* gcn_b  = (const float*)d_in[7];
    const float* mlp_w1 = (const float*)d_in[8];
    const float* mlp_b1 = (const float*)d_in[9];
    const float* mlp_w2 = (const float*)d_in[10];
    const float* mlp_b2 = (const float*)d_in[11];
    const float* ln2_g  = (const float*)d_in[12];
    const float* ln2_b  = (const float*)d_in[13];

    const int N = in_sizes[0] / IN_F;     // 20000
    const int E = in_sizes[1] / 2;        // 100000
    const int* src = ei;
    const int* dst = ei + E;

    // ---- workspace carve (256B-aligned regions)
    char* p = (char*)d_ws;
    auto carve = [&](size_t bytes) -> char* {
        char* r = p;
        p += (bytes + 255) & ~(size_t)255;
        return r;
    };
    float*          cat   = (float*)         carve((size_t)N * CATW * 4);  // fp32 scatter accum
    unsigned short* catb  = (unsigned short*)carve((size_t)N * CATW * 2);  // bf16 activation mirror
    unsigned short* xb    = (unsigned short*)carve((size_t)N * IN_F * 2);
    float*          tmp   = (float*)         carve((size_t)N * H * 4);     // hop GEMM out; reused for mlp2 out
    unsigned short* tmpb  = (unsigned short*)carve((size_t)N * H * 2);     // bf16 z (mlp1 out)
    unsigned short* projT = (unsigned short*)carve((size_t)H * IN_F * 2);
    unsigned short* gcnT  = (unsigned short*)carve((size_t)HOPS * H * H * 2);
    unsigned short* mlp1T = (unsigned short*)carve((size_t)H * CATW * 2);
    unsigned short* mlp2T = (unsigned short*)carve((size_t)H * H * 2);
    float*          deg   = (float*)         carve((size_t)N * 4);
    float*          dis   = (float*)         carve((size_t)N * 4);

    const dim3 blk(256);
    auto nb = [](size_t total) { return (unsigned)((total + 255) / 256); };
    const unsigned GEMM_LDS = 32768;

    // ---- one-time per launch: operand conversion (bf16, weights transposed)
    cvt_f32_bf16_kernel<<<nb((size_t)N * IN_F / 4), blk, 0, stream>>>(x, xb, (size_t)N * IN_F / 4);
    transpose_bf16_kernel<<<nb((size_t)IN_F * H), blk, 0, stream>>>(proj_w, projT, IN_F, H);
    for (int k = 0; k < HOPS; ++k)
        transpose_bf16_kernel<<<nb((size_t)H * H), blk, 0, stream>>>(
            gcn_w + (size_t)k * H * H, gcnT + (size_t)k * H * H, H, H);
    transpose_bf16_kernel<<<nb((size_t)CATW * H), blk, 0, stream>>>(mlp_w1, mlp1T, CATW, H);
    transpose_bf16_kernel<<<nb((size_t)H * H), blk, 0, stream>>>(mlp_w2, mlp2T, H, H);

    // ---- degree normalization
    init_deg_kernel<<<nb(N), blk, 0, stream>>>(deg, N);
    count_deg_kernel<<<nb(E), blk, 0, stream>>>(dst, deg, E);
    make_dis_kernel<<<nb(N), blk, 0, stream>>>(deg, dis, N);

    const dim3 gemm_grid((N + BM - 1) / BM, H / BN);   // (157, 6)
    const unsigned nb_nh = nb((size_t)N * H);

    // ---- proj -> cat slice0 (fp32), then LN1 -> catb slice0 (bf16)
    gemm_bf16_wmma<<<gemm_grid, blk, GEMM_LDS, stream>>>(
        xb, IN_F, projT, IN_F,
        cat, CATW, nullptr, 0, nullptr, 0, nullptr,
        proj_b, N, H, IN_F, 0);
    layernorm_kernel<<<N, blk, 0, stream>>>(
        cat, CATW, nullptr, 0, catb, CATW, ln1_g, ln1_b, H);

    // ---- hops: GEMM (fp32 tmp + self-loop-scaled init of cat slice k+1),
    //            edge scatter-add, then bias+GELU -> catb slice k+1
    for (int k = 0; k < HOPS; ++k) {
        const unsigned short* hk = catb + (size_t)k * H;       // bf16 A slice
        float* hkp1 = cat + (size_t)(k + 1) * H;               // fp32 accum slice
        gemm_bf16_wmma<<<gemm_grid, blk, GEMM_LDS, stream>>>(
            hk, CATW, gcnT + (size_t)k * H * H, H,
            tmp, H, nullptr, 0, hkp1, CATW, dis,
            nullptr, N, H, H, 0);
        edge_scatter_kernel<<<E, blk, 0, stream>>>(src, dst, dis, tmp, hkp1, CATW, E, H);
        gelu_bias_kernel<<<nb_nh, blk, 0, stream>>>(
            hkp1, CATW, catb + (size_t)(k + 1) * H, CATW, gcn_b + (size_t)k * H, N, H);
    }

    // ---- MLP: z = gelu(cat @ w1 + b1) -> bf16 only; out = z @ w2 + b2 -> fp32
    gemm_bf16_wmma<<<gemm_grid, blk, GEMM_LDS, stream>>>(
        catb, CATW, mlp1T, CATW,
        nullptr, 0, tmpb, H, nullptr, 0, nullptr,
        mlp_b1, N, H, CATW, 1 /*fused GELU*/);
    gemm_bf16_wmma<<<gemm_grid, blk, GEMM_LDS, stream>>>(
        tmpb, H, mlp2T, H,
        tmp, H, nullptr, 0, nullptr, 0, nullptr,
        mlp_b2, N, H, H, 0);

    // ---- final LN -> d_out (fp32)
    layernorm_kernel<<<N, blk, 0, stream>>>(
        tmp, H, (float*)d_out, H, nullptr, 0, ln2_g, ln2_b, H);
}
